// LeoModel_21148418966103
// MI455X (gfx1250) — matrix operand order repaired and checked by reading, not compiled, not run on hardware
//
#include <hip/hip_runtime.h>

// ---------------------------------------------------------------------------
// Types for CDNA5 WMMA (wave32): v_wmma_f32_16x16x32_f16
// ---------------------------------------------------------------------------
typedef __attribute__((ext_vector_type(16))) _Float16 v16h;
typedef __attribute__((ext_vector_type(8)))  _Float16 v8h;
typedef __attribute__((ext_vector_type(8)))  float    v8f;

static __device__ __forceinline__ v8f wmma_f16(v16h a, v16h b, v8f c) {
  // D = A(16x32) * B(32x16) + C(16x16), f32 accum
  return __builtin_amdgcn_wmma_f32_16x16x32_f16(
      /*neg_a=*/false, a, /*neg_b=*/false, b,
      /*c_mod=*/(short)0, c, /*reuse_a=*/false, /*reuse_b=*/false);
}

// A-matrix 16x32 f16 fragment from row-major f16 (ld = row stride in halves).
// ISA layout: lane<16 -> row M=lane, K {0..7, 16..23}; lane>=16 -> row M=lane-16,
// K {8..15, 24..31}. Each lane: two contiguous 16B chunks.
static __device__ __forceinline__ v16h load_a(const _Float16* base, int ld) {
  int lane = threadIdx.x & 31;
  int m = lane & 15;
  int koff = (lane >> 4) * 8;
  const _Float16* p = base + m * ld + koff;
  union { v16h v; v8h h[2]; } u;
  u.h[0] = *(const v8h*)(p);        // K: koff .. koff+7
  u.h[1] = *(const v8h*)(p + 16);   // K: koff+16 .. koff+23
  return u.v;
}

// B-matrix 32x16 f16 fragment from PRE-TRANSPOSED weights Bt (N rows x K cols,
// row-major, ld = K stride). ISA layout: lane<16 -> col N=lane, K {0..7, 16..23};
// lane>=16 -> col N=lane-16, K {8..15, 24..31}.
static __device__ __forceinline__ v16h load_bt(const _Float16* bt, int ld,
                                               int n0, int k0) {
  int lane = threadIdx.x & 31;
  int n = lane & 15;
  int kk = (lane >> 4) * 8;
  const _Float16* p = bt + (n0 + n) * ld + k0 + kk;
  union { v16h v; v8h h[2]; } u;
  u.h[0] = *(const v8h*)(p);
  u.h[1] = *(const v8h*)(p + 16);
  return u.v;
}

// ---------------------------------------------------------------------------
// Tensor Data Mover (gfx1250): async 2-D f16 tile load global -> LDS.
// Descriptor per CDNA5 ISA (D# groups 0/1; 2/3 zero).
// Guarded so toolchains without the builtin fall back to plain LDS copies.
// ---------------------------------------------------------------------------
#if defined(__AMDGCN__) && __has_builtin(__builtin_amdgcn_tensor_load_to_lds) && \
    __has_builtin(__builtin_amdgcn_s_wait_tensorcnt)
#define HAVE_TDM 1
#else
#define HAVE_TDM 0
#endif

#if HAVE_TDM
typedef __attribute__((ext_vector_type(4))) unsigned int v4u;
typedef __attribute__((ext_vector_type(8))) int          v8i;
typedef __attribute__((ext_vector_type(4))) int          v4i;

static __device__ __forceinline__ void tdm_load_tile_f16(
    unsigned lds_byte_addr, const _Float16* gsrc,
    unsigned width_elems, unsigned height_rows, unsigned row_stride_elems) {
  unsigned long long ga = (unsigned long long)(size_t)gsrc;
  v4u g0 = {0u, 0u, 0u, 0u};
  g0[0] = 1u;                                    // count=1 (valid user D#)
  g0[1] = lds_byte_addr;                         // D#.lds_addr
  g0[2] = (unsigned)ga;                          // global_addr[31:0]
  g0[3] = ((unsigned)(ga >> 32) & 0x01ffffffu)   // global_addr[56:32]
          | (2u << 30);                          // type = 2 ("image")
  v8i g1 = {0, 0, 0, 0, 0, 0, 0, 0};
  g1[0] = (int)(1u << 16);                       // data_size=1 -> 2 bytes/elem
  g1[1] = (int)((width_elems & 0xffffu) << 16);  // tensor_dim0[15:0]
  g1[2] = (int)((width_elems >> 16) |            // tensor_dim0[31:16]
                ((height_rows & 0xffffu) << 16));// tensor_dim1[15:0]
  g1[3] = (int)((height_rows >> 16) |            // tensor_dim1[31:16]
                ((width_elems & 0xffffu) << 16));// tile_dim0
  g1[4] = (int)(height_rows & 0xffffu);          // tile_dim1 (tile_dim2=0)
  g1[5] = (int)row_stride_elems;                 // tensor_dim0_stride[31:0]
  v4i gz = {0, 0, 0, 0};
#if __has_include(<hip/amd_detail/amd_gfx1250_TDM.h>)
  v8i gz8 = {0, 0, 0, 0, 0, 0, 0, 0};
  __builtin_amdgcn_tensor_load_to_lds(g0, g1, gz, gz, gz8, 0);   // clang-23 form
#else
  __builtin_amdgcn_tensor_load_to_lds(g0, g1, gz, gz, 0);        // ROCm 7.2 form
#endif
}
#endif  // HAVE_TDM

// ---------------------------------------------------------------------------
// Problem constants
// ---------------------------------------------------------------------------
#define EDIM 256
#define HDIM 128
#define MDIM 301
#define BB   16
#define NN   16
#define NKP  128            // n*k pair rows per batch
#define H2   256            // 2*H
#define ROWS 2048           // B*N*K

// Workspace layout (bytes, 256-aligned)
#define OFF_STATS 0u              // 16*16*256 f32      = 262144 B
#define OFF_XH    262144u         // 2048*256 f16       = 1048576 B
#define OFF_F     1310720u        // 16*128*128 f16     = 524288 B
#define OFF_U     1835008u        // 16*128*256 f16     = 1048576 B
#define OFF_V     2883584u        // 16*128*256 f16     = 1048576 B
#define OFF_W1T   3932160u        // 128x256 f16        = 65536 B
#define OFF_W2T   3997696u        // 128x128 f16        = 32768 B
#define OFF_WR1T  4030464u        // 256x128 f16        = 65536 B
#define OFF_WR1B  4096000u        // 256x128 f16        = 65536 B
#define OFF_WR2T  4161536u        // 256x256 f16        = 131072 B
#define OFF_WR3T  4292608u        // 256x256 f16        = 131072 B

// ---------------------------------------------------------------------------
// Prep kernels
// ---------------------------------------------------------------------------
__global__ void zero_f32(float* p, int n) {
  int i = blockIdx.x * blockDim.x + threadIdx.x;
  if (i < n) p[i] = 0.0f;
}

__global__ void cvt_f32_f16(const float* s, _Float16* d, int n) {
  int i = blockIdx.x * blockDim.x + threadIdx.x;
  if (i < n) d[i] = (_Float16)s[i];
}

// d[n*K + k] = (f16) s[k*ld + n]   (builds B^T for contiguous B fragments)
__global__ void tcvt_f32_f16(const float* s, _Float16* d, int K, int N, int ld) {
  int i = blockIdx.x * blockDim.x + threadIdx.x;
  if (i < K * N) {
    int n = i / K, k = i % K;
    d[i] = (_Float16)s[k * ld + n];
  }
}

// ---------------------------------------------------------------------------
// Stage A: f = relu(relu(x@W1 + b1) @ W2 + b2)   -> F (2048 x 128) f16
// ---------------------------------------------------------------------------
__global__ void stageA(const _Float16* __restrict__ xh,
                       const _Float16* __restrict__ W1T, const float* __restrict__ b1,
                       const _Float16* __restrict__ W2T, const float* __restrict__ b2,
                       _Float16* __restrict__ F) {
  __shared__ __align__(16) _Float16 H1[16 * HDIM];   // 4KB
  int tid = threadIdx.x, lane = tid & 31, w = tid >> 5;
  int l15 = lane & 15, rbase = (lane >> 4) * 8;
  int r0 = blockIdx.x * 16;
  const _Float16* A = xh + (size_t)r0 * EDIM;

  for (int t = 0; t < 2; ++t) {
    int n0 = (w * 2 + t) * 16;
    v8f c = {};
    for (int kc = 0; kc < 8; ++kc) {
      v16h a  = load_a(A + kc * 32, EDIM);
      v16h bb = load_bt(W1T, EDIM, n0, kc * 32);
      c = wmma_f16(a, bb, c);
    }
    float bias = b1[n0 + l15];
    int col = n0 + l15;
#pragma unroll
    for (int r = 0; r < 8; ++r) {
      float v = c[r] + bias;
      H1[(rbase + r) * HDIM + col] = (_Float16)(v > 0.f ? v : 0.f);
    }
  }
  __syncthreads();

  for (int t = 0; t < 2; ++t) {
    int n0 = (w * 2 + t) * 16;
    v8f c = {};
    for (int kc = 0; kc < 4; ++kc) {
      v16h a  = load_a(&H1[kc * 32], HDIM);
      v16h bb = load_bt(W2T, HDIM, n0, kc * 32);
      c = wmma_f16(a, bb, c);
    }
    float bias = b2[n0 + l15];
    int col = n0 + l15;
#pragma unroll
    for (int r = 0; r < 8; ++r) {
      float v = c[r] + bias;
      F[(size_t)(r0 + rbase + r) * HDIM + col] = (_Float16)(v > 0.f ? v : 0.f);
    }
  }
}

// ---------------------------------------------------------------------------
// Stage B: U = f @ Wr1_top + br1 ; V = f @ Wr1_bot     (per batch 128x256)
// ---------------------------------------------------------------------------
__global__ void stageB(const _Float16* __restrict__ F,
                       const _Float16* __restrict__ WtT,
                       const _Float16* __restrict__ WbT,
                       const float* __restrict__ br1,
                       _Float16* __restrict__ U, _Float16* __restrict__ V) {
  int mt = blockIdx.x, b = blockIdx.y, sel = blockIdx.z;
  int lane = threadIdx.x & 31, w = threadIdx.x >> 5;
  int l15 = lane & 15, rbase = (lane >> 4) * 8;
  const _Float16* A  = F + (size_t)(b * NKP + mt * 16) * HDIM;
  const _Float16* BT = sel ? WbT : WtT;
  _Float16* OUT = (sel ? V : U) + (size_t)(b * NKP + mt * 16) * H2;

  for (int t = 0; t < 4; ++t) {
    int n0 = (w * 4 + t) * 16;
    v8f c = {};
    for (int kc = 0; kc < 4; ++kc) {
      v16h a  = load_a(A + kc * 32, HDIM);
      v16h bb = load_bt(BT, HDIM, n0, kc * 32);
      c = wmma_f16(a, bb, c);
    }
    float bias = sel ? 0.f : br1[n0 + l15];  // fold br1 into U only
    int col = n0 + l15;
#pragma unroll
    for (int r = 0; r < 8; ++r)
      OUT[(size_t)(rbase + r) * H2 + col] = (_Float16)(c[r] + bias);
  }
}

// ---------------------------------------------------------------------------
// Stage C (hot): pairwise layers 2,3 + row-group sums.
// grid (8 itiles, 16 batches, 4 jsplit) x 128 thr (4 waves).
// Ut / Vt fed by the Tensor Data Mover (double-buffered Vt overlaps the DMA
// with the layer-2/3 WMMA work; TENSORcnt waited only after layer 3).
// ---------------------------------------------------------------------------
__global__ void stageC(const _Float16* __restrict__ U,
                       const _Float16* __restrict__ V,
                       const _Float16* __restrict__ Wr2T, const float* __restrict__ br2,
                       const _Float16* __restrict__ Wr3T, const float* __restrict__ br3,
                       float* __restrict__ stats) {
  __shared__ __align__(16) _Float16 Ut[16 * H2];      //  8KB
  __shared__ __align__(16) _Float16 Vt[2][2 * H2];    //  2KB (double buffer)
  __shared__ __align__(16) _Float16 R1[32 * H2];      // 16KB
  __shared__ __align__(16) _Float16 R2[32 * H2];      // 16KB

  int itile = blockIdx.x, b = blockIdx.y, js = blockIdx.z;
  int tid = threadIdx.x, lane = tid & 31, w = tid >> 5;
  int l15 = lane & 15, hi = lane >> 4, rbase = hi * 8;

  // Warm L2 with the two weight matrices (global_prefetch_b8)
  for (int e = tid; e < 512; e += 128) {
    __builtin_prefetch(Wr2T + e * 128, 0, 3);
    __builtin_prefetch(Wr3T + e * 128, 0, 3);
  }

  const _Float16* Usrc = U + (size_t)(b * NKP + itile * 16) * H2;
  const _Float16* Vbase = V + (size_t)(b * NKP) * H2;

#if HAVE_TDM
  if (tid < 32) {  // wave 0 drives the TDM (EXEC-independent, per-wave op)
    tdm_load_tile_f16((unsigned)(size_t)Ut, Usrc, H2, 16, H2);
    tdm_load_tile_f16((unsigned)(size_t)Vt[0], Vbase + (size_t)(js * 32) * H2,
                      H2, 2, H2);
    __builtin_amdgcn_s_wait_tensorcnt(0);
  }
#else
  {
    const v8h* src = (const v8h*)Usrc;
    v8h* dst = (v8h*)Ut;
    for (int e = tid; e < 512; e += 128) dst[e] = src[e];
  }
#endif

  float acc[4] = {0.f, 0.f, 0.f, 0.f};

  for (int it = 0; it < 16; ++it) {
    int j0 = js * 32 + it * 2;
    __syncthreads();   // TDM/Vt data visible; R1,R2 free from previous iter
#if !HAVE_TDM
    {
      const v8h* src = (const v8h*)(Vbase + (size_t)j0 * H2);
      v8h* dst = (v8h*)Vt[it & 1];
      if (tid < 64) dst[tid] = src[tid];
    }
    __syncthreads();
#endif

    // Build R1 = relu(Ut[ii] + Vt[jj]) ; row = jj*16+ii ; 64 halves/thread
    {
      int row = tid >> 2;
      int c0 = (tid & 3) * 64;
      int jj = row >> 4, ii = row & 15;
      const _Float16* up = Ut + ii * H2 + c0;
      const _Float16* vp = Vt[it & 1] + jj * H2 + c0;
      _Float16* rp = R1 + row * H2 + c0;
#pragma unroll
      for (int c = 0; c < 64; ++c) {
        float v = (float)up[c] + (float)vp[c];
        rp[c] = (_Float16)(v > 0.f ? v : 0.f);
      }
    }
    __syncthreads();

#if HAVE_TDM
    // Vt[it&1] consumed: prefetch next j-pair into the other buffer, DMA
    // overlaps with the layer-2/3 WMMA work below.
    if (tid < 32 && it + 1 < 16) {
      tdm_load_tile_f16((unsigned)(size_t)Vt[(it + 1) & 1],
                        Vbase + (size_t)(j0 + 2) * H2, H2, 2, H2);
    }
#endif

    // Layer 2: R2 = relu(R1 @ Wr2 + br2)   (32x256 @ 256x256)
    for (int t = 0; t < 4; ++t) {
      int n0 = (w * 4 + t) * 16;
      v8f c0v = {}, c1v = {};
      for (int kc = 0; kc < 8; ++kc) {
        int k0 = kc * 32;
        v16h bb = load_bt(Wr2T, H2, n0, k0);
        v16h a0 = load_a(R1 + k0, H2);
        v16h a1 = load_a(R1 + 16 * H2 + k0, H2);
        c0v = wmma_f16(a0, bb, c0v);
        c1v = wmma_f16(a1, bb, c1v);
      }
      float bias = br2[n0 + l15];
      int col = n0 + l15;
#pragma unroll
      for (int r = 0; r < 8; ++r) {
        float v = c0v[r] + bias;
        R2[(rbase + r) * H2 + col] = (_Float16)(v > 0.f ? v : 0.f);
        v = c1v[r] + bias;
        R2[(16 + rbase + r) * H2 + col] = (_Float16)(v > 0.f ? v : 0.f);
      }
    }
    __syncthreads();

    // Layer 3 + relu + per-lane row-group sums (never stored)
    for (int t = 0; t < 4; ++t) {
      int n0 = (w * 4 + t) * 16;
      v8f c0v = {}, c1v = {};
      for (int kc = 0; kc < 8; ++kc) {
        int k0 = kc * 32;
        v16h bb = load_bt(Wr3T, H2, n0, k0);
        v16h a0 = load_a(R2 + k0, H2);
        v16h a1 = load_a(R2 + 16 * H2 + k0, H2);
        c0v = wmma_f16(a0, bb, c0v);
        c1v = wmma_f16(a1, bb, c1v);
      }
      float bias = br3[n0 + l15];
      float s = 0.f;
#pragma unroll
      for (int r = 0; r < 8; ++r) {
        float v = c0v[r] + bias; s += (v > 0.f ? v : 0.f);
        v = c1v[r] + bias;       s += (v > 0.f ? v : 0.f);
      }
      acc[t] += s;
    }

#if HAVE_TDM
    if (tid < 32 && it + 1 < 16)
      __builtin_amdgcn_s_wait_tensorcnt(0);   // next Vt landed before top sync
#endif
  }

  // stats[b][n][c] += acc ; n = itile*2 + (lane>=16)
  int n = itile * 2 + hi;
  float* srow = stats + (size_t)(b * NN + n) * H2;
#pragma unroll
  for (int t = 0; t < 4; ++t)
    atomicAdd(&srow[(w * 4 + t) * 16 + l15], acc[t]);
}

// ---------------------------------------------------------------------------
// Stage D: latents = stats_mean + stats_var*z1 ; out = decode (fp32 exact)
// ---------------------------------------------------------------------------
__global__ void stageD(const float* __restrict__ stats, const float* __restrict__ z1,
                       const float* __restrict__ Wd, const float* __restrict__ bd,
                       const float* __restrict__ z2, float* __restrict__ out) {
  __shared__ float lat[HDIM];
  int bn = blockIdx.x;
  int t = threadIdx.x;
  const float inv = 1.0f / 1024.0f;   // mean over k*nk = 8*128
  lat[t] = stats[(size_t)bn * H2 + t] * inv
         + stats[(size_t)bn * H2 + HDIM + t] * inv * z1[(size_t)bn * HDIM + t];
  __syncthreads();
  for (int m = t; m < MDIM; m += 128) {
    float w1 = bd[m], w2 = bd[MDIM + m];
    for (int h = 0; h < HDIM; ++h) {
      float l = lat[h];
      w1 += l * Wd[(size_t)h * (2 * MDIM) + m];
      w2 += l * Wd[(size_t)h * (2 * MDIM) + MDIM + m];
    }
    out[(size_t)bn * MDIM + m] = w1 + w2 * z2[(size_t)bn * MDIM + m];
  }
}

// ---------------------------------------------------------------------------
extern "C" void kernel_launch(void* const* d_in, const int* in_sizes, int n_in,
                              void* d_out, int out_size, void* d_ws, size_t ws_size,
                              hipStream_t stream) {
  (void)in_sizes; (void)n_in; (void)out_size; (void)ws_size;
  const float* x   = (const float*)d_in[0];
  const float* z1  = (const float*)d_in[1];
  const float* z2  = (const float*)d_in[2];
  const float* W1  = (const float*)d_in[3];
  const float* b1  = (const float*)d_in[4];
  const float* W2  = (const float*)d_in[5];
  const float* b2  = (const float*)d_in[6];
  const float* Wr1 = (const float*)d_in[7];
  const float* br1 = (const float*)d_in[8];
  const float* Wr2 = (const float*)d_in[9];
  const float* br2 = (const float*)d_in[10];
  const float* Wr3 = (const float*)d_in[11];
  const float* br3 = (const float*)d_in[12];
  const float* Wd  = (const float*)d_in[13];
  const float* bd  = (const float*)d_in[14];
  float* out = (float*)d_out;
  char* ws = (char*)d_ws;

  float*    stats = (float*)(ws + OFF_STATS);
  _Float16* xh    = (_Float16*)(ws + OFF_XH);
  _Float16* F     = (_Float16*)(ws + OFF_F);
  _Float16* Uw    = (_Float16*)(ws + OFF_U);
  _Float16* Vw    = (_Float16*)(ws + OFF_V);
  _Float16* W1T   = (_Float16*)(ws + OFF_W1T);
  _Float16* W2T   = (_Float16*)(ws + OFF_W2T);
  _Float16* Wr1tT = (_Float16*)(ws + OFF_WR1T);
  _Float16* Wr1bT = (_Float16*)(ws + OFF_WR1B);
  _Float16* Wr2T  = (_Float16*)(ws + OFF_WR2T);
  _Float16* Wr3T  = (_Float16*)(ws + OFF_WR3T);

  // Prep
  zero_f32<<<(NN * BB * H2 + 255) / 256, 256, 0, stream>>>(stats, NN * BB * H2);
  cvt_f32_f16<<<(ROWS * EDIM + 255) / 256, 256, 0, stream>>>(x, xh, ROWS * EDIM);
  tcvt_f32_f16<<<(256 * 128 + 255) / 256, 256, 0, stream>>>(W1,  W1T,  256, 128, 128);
  tcvt_f32_f16<<<(128 * 128 + 255) / 256, 256, 0, stream>>>(W2,  W2T,  128, 128, 128);
  tcvt_f32_f16<<<(128 * 256 + 255) / 256, 256, 0, stream>>>(Wr1,             Wr1tT, 128, 256, 256);
  tcvt_f32_f16<<<(128 * 256 + 255) / 256, 256, 0, stream>>>(Wr1 + 128 * 256, Wr1bT, 128, 256, 256);
  tcvt_f32_f16<<<(256 * 256 + 255) / 256, 256, 0, stream>>>(Wr2, Wr2T, 256, 256, 256);
  tcvt_f32_f16<<<(256 * 256 + 255) / 256, 256, 0, stream>>>(Wr3, Wr3T, 256, 256, 256);

  // Pipeline
  stageA<<<128, 128, 0, stream>>>(xh, W1T, b1, W2T, b2, F);
  stageB<<<dim3(8, 16, 2), 128, 0, stream>>>(F, Wr1tT, Wr1bT, br1, Uw, Vw);
  stageC<<<dim3(8, 16, 4), 128, 0, stream>>>(Uw, Vw, Wr2T, br2, Wr3T, br3, stats);
  stageD<<<256, 128, 0, stream>>>(stats, z1, Wd, bd, z2, out);
}
